// MinLSTMCell_36120674960012
// MI455X (gfx1250) — matrix operand order repaired and checked by reading, not compiled, not run on hardware
//
#include <hip/hip_runtime.h>
#include <math.h>

typedef __attribute__((ext_vector_type(16))) __bf16        v16bf;
typedef __attribute__((ext_vector_type(8)))  float         v8f;
typedef __attribute__((ext_vector_type(4)))  float         v4f;
typedef __attribute__((ext_vector_type(4)))  unsigned int  v4u;

#define BB     8
#define TT     4096
#define DD     1024
#define HH     1024
#define TP1    4097        // T+1 output timesteps
#define CH_LEN 64          // scan chunk length
#define NCH    65          // ceil(4097/64)

// ---------------- helpers ----------------

static __device__ __forceinline__ unsigned short f2bf(float f) {
  unsigned u = __float_as_uint(f);
  u += 0x7FFFu + ((u >> 16) & 1u);          // round-to-nearest-even
  return (unsigned short)(u >> 16);
}
static __device__ __forceinline__ unsigned pk2(float lo, float hi) {
  return (unsigned)f2bf(lo) | ((unsigned)f2bf(hi) << 16);
}
static __device__ __forceinline__ float softplusf(float x) {
  return fmaxf(x, 0.f) + log1pf(expf(-fabsf(x)));
}
static __device__ __forceinline__ float log_gf(float x) {
  return (x >= 0.f) ? logf(x + 0.5f) : -softplusf(-x);
}
static __device__ __forceinline__ float logaddexpf_(float a, float b) {
  float m = fmaxf(a, b);
  return m + log1pf(expf(fminf(a, b) - m));
}

// ---------------- K1: fused 3-way GEMM (bf16 WMMA) + elementwise epilogue ----------------
// grid (HH/64, (BB*TT)/128), block 256 (8 waves).
// Fully scalarized: no local arrays anywhere -> nothing can fall into scratch.

__global__ __attribute__((amdgpu_waves_per_eu(1))) __launch_bounds__(256)
void k_gemm_epilogue(const float* __restrict__ x,
                     const float* __restrict__ Wf, const float* __restrict__ bfp,
                     const float* __restrict__ Wi, const float* __restrict__ bip,
                     const float* __restrict__ Wh, const float* __restrict__ bhp,
                     float* __restrict__ lfbuf, float* __restrict__ out)
{
  // padded strides: 40 shorts (80 B) per row/col -> 16B-aligned, conflict-free
  __shared__ __align__(16) unsigned short sA[128 * 40];         // A tile 128x32 bf16
  __shared__ __align__(16) unsigned short sB[3 * 64 * 40];      // B tiles [mat][n][k] bf16

  const int tid  = threadIdx.x;
  const int lane = tid & 31;
  const int wv   = tid >> 5;        // wave 0..7 -> M sub-tile
  const int hi   = lane >> 4;       // half-wave select
  const int ln   = lane & 15;

  const int ncol0 = blockIdx.x * 64;
  const int row0  = blockIdx.y * 128;

  const v8f vzero = (v8f){0.f,0.f,0.f,0.f,0.f,0.f,0.f,0.f};
  v8f aF0 = vzero, aF1 = vzero, aF2 = vzero, aF3 = vzero;
  v8f aI0 = vzero, aI1 = vzero, aI2 = vzero, aI3 = vzero;
  v8f aH0 = vzero, aH1 = vzero, aH2 = vzero, aH3 = vzero;

  // staging coordinates
  const int arow  = tid >> 1;       // 0..127
  const int ahalf = tid & 1;        // K half (0..15 / 16..31)
  const int bkp   = tid >> 4;       // K pair index 0..15
  const int bng   = tid & 15;       // N group-of-4 index

  const float* xbase  = x  + (size_t)(row0 + arow) * DD + ahalf * 16;
  const float* wfbase = Wf + (size_t)(2 * bkp) * HH + ncol0 + bng * 4;
  const float* wibase = Wi + (size_t)(2 * bkp) * HH + ncol0 + bng * 4;
  const float* whbase = Wh + (size_t)(2 * bkp) * HH + ncol0 + bng * 4;

  // prefetch registers (all named scalars/vectors -> SSA)
  v4f pa0, pa1, pa2, pa3;
  v4f pf0, pf1, pi0, pi1, ph0, ph1;

#define GLB_LOAD(K0)                                                \
  do {                                                              \
    const float* xg = xbase + (K0);                                 \
    pa0 = *(const v4f*)(xg + 0);                                    \
    pa1 = *(const v4f*)(xg + 4);                                    \
    pa2 = *(const v4f*)(xg + 8);                                    \
    pa3 = *(const v4f*)(xg + 12);                                   \
    const size_t wo = (size_t)(K0) * HH;                            \
    pf0 = *(const v4f*)(wfbase + wo);                               \
    pf1 = *(const v4f*)(wfbase + wo + HH);                          \
    pi0 = *(const v4f*)(wibase + wo);                               \
    pi1 = *(const v4f*)(wibase + wo + HH);                          \
    ph0 = *(const v4f*)(whbase + wo);                               \
    ph1 = *(const v4f*)(whbase + wo + HH);                          \
  } while (0)

#define STORE_B(MI, P0, P1)                                                         \
  do {                                                                              \
    *(unsigned*)(&sB[(MI)*2560 + (bng*4 + 0)*40 + bkp*2]) = pk2((P0)[0], (P1)[0]);  \
    *(unsigned*)(&sB[(MI)*2560 + (bng*4 + 1)*40 + bkp*2]) = pk2((P0)[1], (P1)[1]);  \
    *(unsigned*)(&sB[(MI)*2560 + (bng*4 + 2)*40 + bkp*2]) = pk2((P0)[2], (P1)[2]);  \
    *(unsigned*)(&sB[(MI)*2560 + (bng*4 + 3)*40 + bkp*2]) = pk2((P0)[3], (P1)[3]);  \
  } while (0)

#define WMMA_ONE(ACC, MI, NS)                                                       \
  do {                                                                              \
    union { v4u q[2]; v16bf v; } ub;                                                \
    const unsigned short* bp = &sB[(MI)*2560 + ((NS)*16 + ln)*40 + hi*8];           \
    ub.q[0] = *(const v4u*)(bp);                                                    \
    ub.q[1] = *(const v4u*)(bp + 16);                                               \
    ACC = __builtin_amdgcn_wmma_f32_16x16x32_bf16(                                  \
        false, afr, false, ub.v, (short)0, ACC, false, false);                      \
  } while (0)

  GLB_LOAD(0);

  for (int k0 = 0; k0 < DD; k0 += 32) {
    __syncthreads();   // previous compute must be done reading LDS
    // ---- convert prefetched tile to bf16 and store to LDS ----
    {
      v4u q0 = (v4u){ pk2(pa0[0],pa0[1]), pk2(pa0[2],pa0[3]), pk2(pa1[0],pa1[1]), pk2(pa1[2],pa1[3]) };
      v4u q1 = (v4u){ pk2(pa2[0],pa2[1]), pk2(pa2[2],pa2[3]), pk2(pa3[0],pa3[1]), pk2(pa3[2],pa3[3]) };
      *(v4u*)(&sA[arow * 40 + ahalf * 16])     = q0;
      *(v4u*)(&sA[arow * 40 + ahalf * 16 + 8]) = q1;
    }
    STORE_B(0, pf0, pf1);
    STORE_B(1, pi0, pi1);
    STORE_B(2, ph0, ph1);
    __syncthreads();

    // ---- prefetch next K tile (completes while WMMAs below execute) ----
    if (k0 + 32 < DD) GLB_LOAD(k0 + 32);

    // ---- A fragment (16x32): lanes<16 hold K{0..7,16..23}, lanes>=16 K{8..15,24..31} ----
    v16bf afr;
    {
      union { v4u q[2]; v16bf v; } ua;
      const unsigned short* ap = &sA[(wv * 16 + ln) * 40 + hi * 8];
      ua.q[0] = *(const v4u*)(ap);
      ua.q[1] = *(const v4u*)(ap + 16);
      afr = ua.v;
    }
    // ---- 12 WMMAs: 3 matrices x 4 N sub-tiles ----
    WMMA_ONE(aF0, 0, 0); WMMA_ONE(aF1, 0, 1); WMMA_ONE(aF2, 0, 2); WMMA_ONE(aF3, 0, 3);
    WMMA_ONE(aI0, 1, 0); WMMA_ONE(aI1, 1, 1); WMMA_ONE(aI2, 1, 2); WMMA_ONE(aI3, 1, 3);
    WMMA_ONE(aH0, 2, 0); WMMA_ONE(aH1, 2, 1); WMMA_ONE(aH2, 2, 2); WMMA_ONE(aH3, 2, 3);
  }

  // ---- epilogue: bias + minLSTM log-space math ----
#define EPILOGUE(NS, AF, AI, AH)                                              \
  do {                                                                        \
    const int n = ncol0 + (NS) * 16 + ln;                                     \
    const float vbf = bfp[n], vbi = bip[n], vbh = bhp[n];                     \
    _Pragma("unroll")                                                         \
    for (int r = 0; r < 8; ++r) {                                             \
      const int mg = row0 + wv * 16 + r + hi * 8; /* mg = b*T + t */          \
      float zf = AF[r] + vbf;                                                 \
      float zi = AI[r] + vbi;                                                 \
      float zh = AH[r] + vbh;                                                 \
      float diff = softplusf(-zf) - softplusf(-zi);                           \
      float lf   = -softplusf(diff);                                          \
      float li   = -softplusf(-diff);                                         \
      float lth  = log_gf(zh);                                                \
      lfbuf[(size_t)mg * HH + n] = lf;                                        \
      out[(size_t)(mg + (mg >> 12) + 1) * HH + n] = li + lth;                 \
    }                                                                         \
  } while (0)

  EPILOGUE(0, aF0, aI0, aH0);
  EPILOGUE(1, aF1, aI1, aH1);
  EPILOGUE(2, aF2, aI2, aH2);
  EPILOGUE(3, aF3, aI3, aH3);

#undef GLB_LOAD
#undef STORE_B
#undef WMMA_ONE
#undef EPILOGUE
}

// ---------------- K2: per-chunk logsumexp of c[t] = val[t] - a_star[t] ----------------

__global__ __launch_bounds__(256)
void k_chunk_reduce(const float* __restrict__ lfbuf, const float* __restrict__ out,
                    const float* __restrict__ hprev, float* __restrict__ seg)
{
  const int idx  = blockIdx.x * 256 + threadIdx.x;   // (b, c, h), h fastest
  const int h    = idx & (HH - 1);
  const int rest = idx >> 10;
  const int c    = rest % NCH;
  const int b    = rest / NCH;
  const int t0 = c * CH_LEN;
  int t1 = t0 + CH_LEN; if (t1 > TP1) t1 = TP1;
  float r = -__builtin_inff();
  for (int t = t0; t < t1; ++t) {
    float astar, val;
    if (t == 0) {
      astar = 0.f;
      val = log_gf(hprev[b * HH + h]);
    } else {
      astar = lfbuf[((size_t)b * TT + (t - 1)) * HH + h];
      val   = out[((size_t)b * TP1 + t) * HH + h];
    }
    r = logaddexpf_(r, val - astar);
  }
  seg[idx] = r;
}

// ---------------- K3: exclusive log-prefix over chunks (per channel) ----------------

__global__ __launch_bounds__(256)
void k_chunk_prefix(const float* __restrict__ seg, float* __restrict__ pre)
{
  const int idx = blockIdx.x * 256 + threadIdx.x;    // b*HH + h
  const int h = idx & (HH - 1);
  const int b = idx >> 10;
  float r = -__builtin_inff();
  for (int c = 0; c < NCH; ++c) {
    const size_t o = ((size_t)b * NCH + c) * HH + h;
    pre[o] = r;
    r = logaddexpf_(r, seg[o]);
  }
}

// ---------------- K4: chunk rescan with carry, emit exp(a_star + cumlse) ----------------
// Reads lv from out slot t, then overwrites the same slot (same thread, read-before-write).

__global__ __launch_bounds__(256)
void k_chunk_scan(const float* __restrict__ lfbuf, const float* __restrict__ hprev,
                  const float* __restrict__ pre, float* __restrict__ out)
{
  const int idx  = blockIdx.x * 256 + threadIdx.x;
  const int h    = idx & (HH - 1);
  const int rest = idx >> 10;
  const int c    = rest % NCH;
  const int b    = rest / NCH;
  const int t0 = c * CH_LEN;
  int t1 = t0 + CH_LEN; if (t1 > TP1) t1 = TP1;
  float r = pre[idx];
  for (int t = t0; t < t1; ++t) {
    float astar, val;
    if (t == 0) {
      astar = 0.f;
      val = log_gf(hprev[b * HH + h]);
    } else {
      astar = lfbuf[((size_t)b * TT + (t - 1)) * HH + h];
      val   = out[((size_t)b * TP1 + t) * HH + h];
    }
    r = logaddexpf_(r, val - astar);
    out[((size_t)b * TP1 + t) * HH + h] = expf(astar + r);
  }
}

// ---------------- launch ----------------

extern "C" void kernel_launch(void* const* d_in, const int* in_sizes, int n_in,
                              void* d_out, int out_size, void* d_ws, size_t ws_size,
                              hipStream_t stream)
{
  (void)in_sizes; (void)n_in; (void)out_size; (void)ws_size;
  const float* x  = (const float*)d_in[0];
  const float* hp = (const float*)d_in[1];
  const float* Wf = (const float*)d_in[2];
  const float* bf = (const float*)d_in[3];
  const float* Wi = (const float*)d_in[4];
  const float* bi = (const float*)d_in[5];
  const float* Wh = (const float*)d_in[6];
  const float* bh = (const float*)d_in[7];
  float* out = (float*)d_out;

  float* lfbuf = (float*)d_ws;                               // B*T*H floats
  float* seg   = lfbuf + (size_t)BB * TT * HH;               // B*NCH*H floats
  float* pre   = seg   + (size_t)BB * NCH * HH;              // B*NCH*H floats

  dim3 g1(HH / 64, (BB * TT) / 128);
  k_gemm_epilogue<<<g1, 256, 0, stream>>>(x, Wf, bf, Wi, bi, Wh, bh, lfbuf, out);

  const int nscan = BB * NCH * HH;                           // 532480
  k_chunk_reduce<<<nscan / 256, 256, 0, stream>>>(lfbuf, out, hp, seg);
  k_chunk_prefix<<<(BB * HH) / 256, 256, 0, stream>>>(seg, pre);
  k_chunk_scan  <<<nscan / 256, 256, 0, stream>>>(lfbuf, hp, pre, out);
}
